// LearnableFusionV4_73658689126443
// MI455X (gfx1250) — matrix operand order repaired and checked by reading, not compiled, not run on hardware
//
#include <hip/hip_runtime.h>
#include <hip/hip_bf16.h>
#include <math.h>

typedef __attribute__((ext_vector_type(16))) _Float16 v16h;
typedef __attribute__((ext_vector_type(8)))  float    v8f;

#define HID 64
#define HEADS 4
#define HD 16
#define EDGE_DIM 6
#define KCLUST 25
#define NEG_SLOPE 0.2f
#define TEMP 0.1f

// ---------------------------------------------------------------------------
// Generic fill
// ---------------------------------------------------------------------------
__global__ void fill_kernel(float* __restrict__ p, float v, int n) {
    int i = blockIdx.x * blockDim.x + threadIdx.x;
    if (i < n) p[i] = v;
}

// ---------------------------------------------------------------------------
// Build padded node features nf32[N][32] = [x(4), pos(3), type_emb(8), 0...]
// (zero-padded so the WMMA GEMM needs no K bounds checks)
// ---------------------------------------------------------------------------
__global__ void build_nf_kernel(const float* __restrict__ x,
                                const float* __restrict__ pos,
                                const int*   __restrict__ node_type,
                                const float* __restrict__ type_emb,
                                float* __restrict__ nf, int N) {
    int n = blockIdx.x * blockDim.x + threadIdx.x;
    if (n >= N) return;
    float* r = nf + (size_t)n * 32;
    r[0] = x[n * 4 + 0]; r[1] = x[n * 4 + 1]; r[2] = x[n * 4 + 2]; r[3] = x[n * 4 + 3];
    r[4] = pos[n * 3 + 0]; r[5] = pos[n * 3 + 1]; r[6] = pos[n * 3 + 2];
    int t = node_type[n];
    #pragma unroll
    for (int k = 0; k < 8; ++k) r[7 + k] = type_emb[t * 8 + k];
    #pragma unroll
    for (int k = 15; k < 32; ++k) r[k] = 0.f;
}

// Pad W[rows,64] -> Wp[rowsPad,64] with zero rows
__global__ void pad_w_kernel(const float* __restrict__ W, float* __restrict__ Wp,
                             int rows, int rowsPad) {
    int i = blockIdx.x * blockDim.x + threadIdx.x;
    if (i >= rowsPad * 64) return;
    int r = i >> 6;
    Wp[i] = (r < rows) ? W[r * 64 + (i & 63)] : 0.f;
}

// ---------------------------------------------------------------------------
// WMMA GEMM: out[M,64] = act(A[M, Kc*32] @ W[Kc*32, 64] + bias)
// Block = 256 threads (8 waves), each wave computes one 16x64 tile.
// grid.x = M/128.  No bounds checks: K padded to a multiple of 32.
//
// A layout used here (16-bit A 16x32, same k order as B): M = lane&15,
//   element e <-> K = half*16 + e               (contiguous)
//   -> four float4 global loads per chunk.
// B layout (32x16): N = lane&15, element e <-> K = half*16 + e
//   -> staged once per block into LDS as f16 in [ch][half][col][e] order,
//      so each lane reads its v16h as one contiguous 32B LDS access.
// D layout (f32 16x16): M = r + 8*half, N = lane&15.
// ---------------------------------------------------------------------------
__global__ __launch_bounds__(256)
void wmma_gemm64_kernel(const float* __restrict__ A, int lda,
                        const float* __restrict__ W, int Kc,
                        const float* __restrict__ bias,
                        float* __restrict__ out, int act) {
    __shared__ __align__(32) _Float16 ldsB[2 * 2 * 64 * 16];  // up to K=64: 8KB

    int t = threadIdx.x;
    // ---- cooperative stage of W into LDS (coalesced: adjacent t -> adjacent col)
    {
        int col = t & 63;
        int kr0 = t >> 6;                       // 0..3
        for (int krow = kr0; krow < Kc * 32; krow += 4) {
            int ch = krow >> 5, hf = (krow >> 4) & 1, e = krow & 15;
            ldsB[(((ch * 2 + hf) * 64) + col) * 16 + e] = (_Float16)W[krow * 64 + col];
        }
    }
    __syncthreads();

    int lane = t & 31;
    int wave = t >> 5;
    int tile = blockIdx.x * 8 + wave;           // 16-row tile index
    int rc   = lane & 15;
    int half = lane >> 4;

    const float* Arow = A + (size_t)(tile * 16 + rc) * lda + half * 16;

    v8f acc[4] = {{}, {}, {}, {}};
    for (int ch = 0; ch < Kc; ++ch) {
        const float* Ap = Arow + ch * 32;
        float4 a0 = *(const float4*)(Ap + 0);
        float4 a1 = *(const float4*)(Ap + 4);
        float4 a2 = *(const float4*)(Ap + 8);
        float4 a3 = *(const float4*)(Ap + 12);
        v16h a;
        a[0]  = (_Float16)a0.x; a[1]  = (_Float16)a0.y; a[2]  = (_Float16)a0.z; a[3]  = (_Float16)a0.w;
        a[4]  = (_Float16)a1.x; a[5]  = (_Float16)a1.y; a[6]  = (_Float16)a1.z; a[7]  = (_Float16)a1.w;
        a[8]  = (_Float16)a2.x; a[9]  = (_Float16)a2.y; a[10] = (_Float16)a2.z; a[11] = (_Float16)a2.w;
        a[12] = (_Float16)a3.x; a[13] = (_Float16)a3.y; a[14] = (_Float16)a3.z; a[15] = (_Float16)a3.w;
        #pragma unroll
        for (int nt = 0; nt < 4; ++nt) {
            int col = nt * 16 + rc;
            const v16h* bp = (const v16h*)&ldsB[(((ch * 2 + half) * 64) + col) * 16];
            v16h b = *bp;
            acc[nt] = __builtin_amdgcn_wmma_f32_16x16x32_f16(false, a, false, b,
                                                             (short)0, acc[nt], false, false);
        }
    }
    #pragma unroll
    for (int nt = 0; nt < 4; ++nt) {
        int col = nt * 16 + rc;
        float bv = bias ? bias[col] : 0.f;
        #pragma unroll
        for (int r = 0; r < 8; ++r) {
            int mrow = r + half * 8;
            float v = acc[nt][r] + bv;
            if (act) v = fmaxf(v, 0.f);
            out[(size_t)(tile * 16 + mrow) * 64 + col] = v;
        }
    }
}

// ---------------------------------------------------------------------------
// Self-loop attr = per-target mean of incoming edge_attr: accumulate cnt & sums
// ---------------------------------------------------------------------------
__global__ void loop_attr_accum_kernel(const int* __restrict__ ei, int E,
                                       const float* __restrict__ edge_attr,
                                       float* __restrict__ cnt, float* __restrict__ asum) {
    int e = blockIdx.x * blockDim.x + threadIdx.x;
    if (e >= E) return;
    int d = ei[E + e];
    atomicAdd(&cnt[d], 1.f);
    #pragma unroll
    for (int k = 0; k < EDGE_DIM; ++k)
        atomicAdd(&asum[d * EDGE_DIM + k], edge_attr[e * EDGE_DIM + k]);
}

// ---------------------------------------------------------------------------
// Float atomic max (int/uint punning trick)
// ---------------------------------------------------------------------------
__device__ __forceinline__ void atomicMaxFloat(float* addr, float val) {
    if (val >= 0.f) atomicMax((int*)addr, __float_as_int(val));
    else            atomicMin((unsigned int*)addr, __float_as_uint(val));
}

// ---------------------------------------------------------------------------
// GATv2 pass 1: per-edge logits (fused ea@we) + segment max via atomics
// ---------------------------------------------------------------------------
__global__ void gat_logits_kernel(const int* __restrict__ ei, int E, int N,
                                  const float* __restrict__ xl,
                                  const float* __restrict__ xr,
                                  const float* __restrict__ edge_attr,
                                  const float* __restrict__ cnt,
                                  const float* __restrict__ asum,
                                  const float* __restrict__ we,   // [6,64]
                                  const float* __restrict__ att,  // [4,16]
                                  float* __restrict__ logits,     // [E+N,4]
                                  float* __restrict__ m) {        // [N,4]
    int e = blockIdx.x * blockDim.x + threadIdx.x;
    int Et = E + N;
    if (e >= Et) return;
    int s, d; float ea[EDGE_DIM];
    if (e < E) {
        s = ei[e]; d = ei[E + e];
        #pragma unroll
        for (int k = 0; k < EDGE_DIM; ++k) ea[k] = edge_attr[e * EDGE_DIM + k];
    } else {
        int i = e - E; s = i; d = i;
        float c = fmaxf(cnt[i], 1.f);
        #pragma unroll
        for (int k = 0; k < EDGE_DIM; ++k) ea[k] = asum[i * EDGE_DIM + k] / c;
    }
    const float* xls = xl + (size_t)s * 64;
    const float* xrd = xr + (size_t)d * 64;
    float lg[HEADS] = {0.f, 0.f, 0.f, 0.f};
    for (int n = 0; n < 64; ++n) {
        float ew = 0.f;
        #pragma unroll
        for (int k = 0; k < EDGE_DIM; ++k) ew += ea[k] * we[k * 64 + n];
        float v = xls[n] + xrd[n] + ew;
        v = v > 0.f ? v : NEG_SLOPE * v;            // leaky_relu
        int h = n >> 4;
        lg[h] += v * att[h * HD + (n & 15)];
    }
    #pragma unroll
    for (int h = 0; h < HEADS; ++h) {
        logits[(size_t)e * 4 + h] = lg[h];
        atomicMaxFloat(&m[d * 4 + h], lg[h]);
    }
}

// ---------------------------------------------------------------------------
// GATv2 pass 2: expa (stored in-place over logits) + segment sum of denom
// ---------------------------------------------------------------------------
__global__ void gat_expsum_kernel(const int* __restrict__ ei, int E, int N,
                                  float* __restrict__ logits,
                                  const float* __restrict__ m,
                                  float* __restrict__ denom) {
    int e = blockIdx.x * blockDim.x + threadIdx.x;
    int Et = E + N;
    if (e >= Et) return;
    int d = (e < E) ? ei[E + e] : (e - E);
    #pragma unroll
    for (int h = 0; h < HEADS; ++h) {
        float x = __expf(logits[(size_t)e * 4 + h] - m[d * 4 + h]);
        logits[(size_t)e * 4 + h] = x;
        atomicAdd(&denom[d * 4 + h], x);
    }
}

// ---------------------------------------------------------------------------
// GATv2 pass 3: alpha-weighted message scatter
// ---------------------------------------------------------------------------
__global__ void gat_scatter_kernel(const int* __restrict__ ei, int E, int N,
                                   const float* __restrict__ xl,
                                   const float* __restrict__ expa,
                                   const float* __restrict__ denom,
                                   float* __restrict__ out) {
    int e = blockIdx.x * blockDim.x + threadIdx.x;
    int Et = E + N;
    if (e >= Et) return;
    int s, d;
    if (e < E) { s = ei[e]; d = ei[E + e]; } else { s = e - E; d = s; }
    float alpha[HEADS];
    #pragma unroll
    for (int h = 0; h < HEADS; ++h)
        alpha[h] = expa[(size_t)e * 4 + h] / denom[d * 4 + h];
    const float* xls = xl + (size_t)s * 64;
    float* od = out + (size_t)d * 64;
    for (int n = 0; n < 64; ++n)
        atomicAdd(&od[n], xls[n] * alpha[n >> 4]);
}

__global__ void gat_finalize_kernel(float* __restrict__ out,
                                    const float* __restrict__ bias, int N, int act) {
    int i = blockIdx.x * blockDim.x + threadIdx.x;
    if (i >= N * 64) return;
    float v = out[i] + bias[i & 63];
    if (act) v = fmaxf(v, 0.f);
    out[i] = v;
}

// ---------------------------------------------------------------------------
// Association matrix: A[i,j] = sigmoid( sum_k w2[k]*relu(si[i,k]+sj[j,k]+b1[k]) + b2 )
// 16x16 output tile per block, si/sj tiles staged through LDS.
// ---------------------------------------------------------------------------
__global__ void assoc_kernel(const float* __restrict__ si,
                             const float* __restrict__ sj,
                             const float* __restrict__ w2,
                             const float* __restrict__ b1,
                             const float* __restrict__ b2p,
                             float* __restrict__ A, int N) {
    __shared__ float ssi[16][64];
    __shared__ float ssj[16][64];
    __shared__ float sw2[64];
    __shared__ float sb1[64];
    int ti = blockIdx.y * 16, tj = blockIdx.x * 16;
    int t = threadIdx.x;  // 256
    {
        // vectorized cooperative loads: 16 rows x 16 float4
        int r = t >> 4, c4 = (t & 15) * 4;
        *(float4*)&ssi[r][c4] = *(const float4*)(si + (size_t)(ti + r) * 64 + c4);
        *(float4*)&ssj[r][c4] = *(const float4*)(sj + (size_t)(tj + r) * 64 + c4);
    }
    if (t < 64) { sw2[t] = w2[t]; sb1[t] = b1[t]; }
    __syncthreads();
    int i = t >> 4, j = t & 15;
    __builtin_prefetch(si + (size_t)(ti + 16) * 64, 0, 0);
    float acc = 0.f;
    #pragma unroll 8
    for (int k = 0; k < 64; ++k) {
        float v = ssi[i][k] + ssj[j][k] + sb1[k];
        acc = fmaf(fmaxf(v, 0.f), sw2[k], acc);
    }
    acc += b2p[0];
    A[(size_t)(ti + i) * N + tj + j] = 1.f / (1.f + __expf(-acc));
}

// ---------------------------------------------------------------------------
// Deterministic farthest-point cluster seeding (argmax variant of kmeans++;
// JAX PRNG categorical sampling is not reproducible device-side).
// ---------------------------------------------------------------------------
__global__ void cluster_select_kernel(const float* __restrict__ A,
                                      float* __restrict__ min_d,
                                      int* __restrict__ idx, int N) {
    __shared__ float sval[256];
    __shared__ int   sidx[256];
    int t = threadIdx.x;
    const int first = 0;
    for (int n = t; n < N; n += 256) min_d[n] = 1.f - A[(size_t)first * N + n];
    if (t == 0) idx[0] = first;
    __syncthreads();
    for (int c = 1; c < KCLUST; ++c) {
        float bv = -1e30f; int bi = 0;
        for (int n = t; n < N; n += 256) {
            float v = min_d[n];
            if (v > bv) { bv = v; bi = n; }
        }
        sval[t] = bv; sidx[t] = bi;
        __syncthreads();
        for (int s = 128; s > 0; s >>= 1) {
            if (t < s && sval[t + s] > sval[t]) { sval[t] = sval[t + s]; sidx[t] = sidx[t + s]; }
            __syncthreads();
        }
        int nxt = sidx[0];
        if (t == 0) idx[c] = nxt;
        __syncthreads();
        for (int n = t; n < N; n += 256)
            min_d[n] = fminf(min_d[n], 1.f - A[(size_t)nxt * N + n]);
        __syncthreads();
    }
}

// ---------------------------------------------------------------------------
// assign[n,c] = softmax_c( A[n, idx[c]] / TEMP )
// ---------------------------------------------------------------------------
__global__ void assign_kernel(const float* __restrict__ A,
                              const int* __restrict__ idx,
                              float* __restrict__ assign, int N) {
    int n = blockIdx.x * blockDim.x + threadIdx.x;
    if (n >= N) return;
    float v[KCLUST], mx = -1e30f;
    #pragma unroll
    for (int c = 0; c < KCLUST; ++c) {
        v[c] = A[(size_t)n * N + idx[c]] * (1.f / TEMP);
        mx = fmaxf(mx, v[c]);
    }
    float s = 0.f;
    #pragma unroll
    for (int c = 0; c < KCLUST; ++c) { v[c] = __expf(v[c] - mx); s += v[c]; }
    float inv = 1.f / s;
    #pragma unroll
    for (int c = 0; c < KCLUST; ++c) assign[(size_t)n * KCLUST + c] = v[c] * inv;
}

// ---------------------------------------------------------------------------
// feats[c,:] = sum_n assign[n,c] * h2[n,:]   (LDS partials, then global atomics)
// ---------------------------------------------------------------------------
__global__ void feats_kernel(const float* __restrict__ assign,
                             const float* __restrict__ h2,
                             float* __restrict__ feats, int N) {
    __shared__ float lf[KCLUST * 64];
    int t = threadIdx.x;
    for (int i = t; i < KCLUST * 64; i += 256) lf[i] = 0.f;
    __syncthreads();
    int n = blockIdx.x * 256 + t;
    if (n < N) {
        const float* hn = h2 + (size_t)n * 64;
        #pragma unroll
        for (int c = 0; c < KCLUST; ++c) {
            float a = assign[(size_t)n * KCLUST + c];
            for (int dd = 0; dd < 64; ++dd)
                atomicAdd(&lf[c * 64 + dd], a * hn[dd]);   // ds_add_f32
        }
    }
    __syncthreads();
    for (int i = t; i < KCLUST * 64; i += 256) atomicAdd(&feats[i], lf[i]);
}

// ---------------------------------------------------------------------------
// Decoder: preds = relu(feats@w1+b1)@w2+b2   [25,64]->[25,7], single block
// ---------------------------------------------------------------------------
__global__ void decoder_kernel(const float* __restrict__ feats,
                               const float* __restrict__ w1, const float* __restrict__ b1,
                               const float* __restrict__ w2, const float* __restrict__ b2,
                               float* __restrict__ preds) {
    __shared__ float hid[KCLUST * 64];
    int t = threadIdx.x;
    for (int i = t; i < KCLUST * 64; i += 256) {
        int r = i >> 6, c = i & 63;
        float s = b1[c];
        for (int k = 0; k < 64; ++k) s = fmaf(feats[r * 64 + k], w1[k * 64 + c], s);
        hid[i] = fmaxf(s, 0.f);
    }
    __syncthreads();
    for (int i = t; i < KCLUST * 7; i += 256) {
        int r = i / 7, o = i % 7;
        float s = b2[o];
        for (int k = 0; k < 64; ++k) s = fmaf(hid[r * 64 + k], w2[k * 7 + o], s);
        preds[i] = s;
    }
}

// ---------------------------------------------------------------------------
// Host launch
// ---------------------------------------------------------------------------
extern "C" void kernel_launch(void* const* d_in, const int* in_sizes, int n_in,
                              void* d_out, int out_size, void* d_ws, size_t ws_size,
                              hipStream_t stream) {
    const float* x         = (const float*)d_in[0];
    const float* pos       = (const float*)d_in[1];
    const int*   node_type = (const int*)  d_in[2];
    const int*   ei        = (const int*)  d_in[3];
    const float* edge_attr = (const float*)d_in[4];
    const float* type_emb  = (const float*)d_in[5];
    const float* enc_w1 = (const float*)d_in[6];  const float* enc_b1 = (const float*)d_in[7];
    const float* enc_w2 = (const float*)d_in[8];  const float* enc_b2 = (const float*)d_in[9];
    const float* g_wl[2]   = {(const float*)d_in[10], (const float*)d_in[17]};
    const float* g_bl[2]   = {(const float*)d_in[11], (const float*)d_in[18]};
    const float* g_wr[2]   = {(const float*)d_in[12], (const float*)d_in[19]};
    const float* g_br[2]   = {(const float*)d_in[13], (const float*)d_in[20]};
    const float* g_we[2]   = {(const float*)d_in[14], (const float*)d_in[21]};
    const float* g_att[2]  = {(const float*)d_in[15], (const float*)d_in[22]};
    const float* g_bias[2] = {(const float*)d_in[16], (const float*)d_in[23]};
    const float* ah_w1 = (const float*)d_in[24]; const float* ah_b1 = (const float*)d_in[25];
    const float* ah_w2 = (const float*)d_in[26]; const float* ah_b2 = (const float*)d_in[27];
    const float* dec_w1 = (const float*)d_in[28]; const float* dec_b1 = (const float*)d_in[29];
    const float* dec_w2 = (const float*)d_in[30]; const float* dec_b2 = (const float*)d_in[31];

    const int N = in_sizes[0] / 4;
    const int E = in_sizes[3] / 2;
    const int Et = E + N;

    // workspace layout (floats)
    float* w = (float*)d_ws;
    float* nf32  = w;                       // N*32 (zero-padded K)
    float* wpad  = nf32 + (size_t)N * 32;   // 32*64 padded enc_w1
    float* tmp   = wpad + 32 * 64;          // N*64
    float* h     = tmp  + (size_t)N * 64;   // N*64
    float* h1    = h    + (size_t)N * 64;   // N*64
    float* h2    = h1   + (size_t)N * 64;   // N*64
    float* xl    = h2   + (size_t)N * 64;   // N*64
    float* xr    = xl   + (size_t)N * 64;   // N*64
    float* si    = xr   + (size_t)N * 64;   // N*64
    float* sj    = si   + (size_t)N * 64;   // N*64
    float* cnt   = sj   + (size_t)N * 64;   // N
    float* asum  = cnt  + N;                // N*6
    float* mmax  = asum + (size_t)N * 6;    // N*4
    float* denom = mmax + (size_t)N * 4;    // N*4
    float* logits= denom+ (size_t)N * 4;    // (E+N)*4
    float* min_d = logits + (size_t)Et * 4; // N
    float* feats = min_d + N;               // 25*64
    int*   idx   = (int*)(feats + KCLUST * 64);

    // output layout: preds[25*7] | A[N*N] | assign[N*25]
    float* out_preds  = (float*)d_out;
    float* A          = out_preds + KCLUST * 7;
    float* out_assign = A + (size_t)N * N;

    const int B = 256;
    auto blocks = [](int n, int b) { return (n + b - 1) / b; };

    // 1) node features + encoder MLP (WMMA GEMMs)
    build_nf_kernel<<<blocks(N, B), B, 0, stream>>>(x, pos, node_type, type_emb, nf32, N);
    pad_w_kernel<<<blocks(32 * 64, B), B, 0, stream>>>(enc_w1, wpad, 15, 32);
    wmma_gemm64_kernel<<<N / 128, 256, 0, stream>>>(nf32, 32, wpad, 1, enc_b1, tmp, 1);
    wmma_gemm64_kernel<<<N / 128, 256, 0, stream>>>(tmp, 64, enc_w2, 2, enc_b2, h, 0);

    // 2) self-loop edge features (mean incoming edge_attr)
    fill_kernel<<<blocks(N * 7, B), B, 0, stream>>>(cnt, 0.f, N * 7); // cnt + asum contiguous
    loop_attr_accum_kernel<<<blocks(E, B), B, 0, stream>>>(ei, E, edge_attr, cnt, asum);

    // 3) two GATv2 layers
    const float* hin = h;
    float* houts[2] = {h1, h2};
    for (int L = 0; L < 2; ++L) {
        float* hout = houts[L];
        wmma_gemm64_kernel<<<N / 128, 256, 0, stream>>>(hin, 64, g_wl[L], 2, g_bl[L], xl, 0);
        wmma_gemm64_kernel<<<N / 128, 256, 0, stream>>>(hin, 64, g_wr[L], 2, g_br[L], xr, 0);
        fill_kernel<<<blocks(N * 4, B), B, 0, stream>>>(mmax, -INFINITY, N * 4);
        fill_kernel<<<blocks(N * 4, B), B, 0, stream>>>(denom, 0.f, N * 4);
        fill_kernel<<<blocks(N * 64, B), B, 0, stream>>>(hout, 0.f, N * 64);
        gat_logits_kernel<<<blocks(Et, B), B, 0, stream>>>(ei, E, N, xl, xr, edge_attr,
                                                           cnt, asum, g_we[L], g_att[L],
                                                           logits, mmax);
        gat_expsum_kernel<<<blocks(Et, B), B, 0, stream>>>(ei, E, N, logits, mmax, denom);
        gat_scatter_kernel<<<blocks(Et, B), B, 0, stream>>>(ei, E, N, xl, logits, denom, hout);
        gat_finalize_kernel<<<blocks(N * 64, B), B, 0, stream>>>(hout, g_bias[L], N, L == 0 ? 1 : 0);
        hin = hout;
    }

    // 4) association matrix (si/sj via WMMA, pairwise VALU kernel)
    wmma_gemm64_kernel<<<N / 128, 256, 0, stream>>>(h2, 64, ah_w1, 2, nullptr, si, 0);
    wmma_gemm64_kernel<<<N / 128, 256, 0, stream>>>(h2, 64, ah_w1 + 64 * 64, 2, nullptr, sj, 0);
    {
        dim3 g(N / 16, N / 16);
        assoc_kernel<<<g, 256, 0, stream>>>(si, sj, ah_w2, ah_b1, ah_b2, A, N);
    }

    // 5) clustering: seed selection, soft assignment, cluster features
    cluster_select_kernel<<<1, 256, 0, stream>>>(A, min_d, idx, N);
    assign_kernel<<<blocks(N, B), B, 0, stream>>>(A, idx, out_assign, N);
    fill_kernel<<<blocks(KCLUST * 64, B), B, 0, stream>>>(feats, 0.f, KCLUST * 64);
    feats_kernel<<<blocks(N, 256), 256, 0, stream>>>(out_assign, h2, feats, N);

    // 6) decoder
    decoder_kernel<<<1, 256, 0, stream>>>(feats, dec_w1, dec_b1, dec_w2, dec_b2, out_preds);
}
